// ParakeetGPT_82386062672033
// MI455X (gfx1250) — matrix operand order repaired and checked by reading, not compiled, not run on hardware
//
#include <hip/hip_runtime.h>
#include <hip/hip_bf16.h>

// ---------------------------------------------------------------------------
// CDNA5 (gfx1250) GPT forward: bf16 WMMA GEMMs + flash attention + f32 head.
// GEMM uses double-buffered GLOBAL_LOAD_ASYNC_TO_LDS_B128 (ASYNCcnt pipeline).
// ---------------------------------------------------------------------------

typedef __attribute__((ext_vector_type(16))) __bf16 v16bf;
typedef __attribute__((ext_vector_type(8)))  float  v8f;
typedef __attribute__((ext_vector_type(4)))  int    v4i;

union FragU { unsigned u[8]; v16bf v; };

#define AS1 __attribute__((address_space(1)))
#define AS3 __attribute__((address_space(3)))

#if __has_builtin(__builtin_amdgcn_global_load_async_to_lds_b128) && \
    __has_builtin(__builtin_amdgcn_s_wait_asynccnt)
#define USE_ASYNC_LDS 1
// async copy: 16B per lane, global -> LDS, tracked by ASYNCcnt (in-order retire)
#define ASYNC_CP16(gp, lp)                                              \
    __builtin_amdgcn_global_load_async_to_lds_b128(                     \
        (AS1 v4i*)(AS1 void*)(void*)(gp),                               \
        (AS3 v4i*)(AS3 void*)(void*)(lp), 0, 0)
#else
#define USE_ASYNC_LDS 0
#endif

static __device__ __forceinline__ __bf16 f2bf(float f) {
    unsigned u = __builtin_bit_cast(unsigned, f);
    unsigned r = (u + 0x7FFFu + ((u >> 16) & 1u)) >> 16;   // round-to-nearest-even
    return __builtin_bit_cast(__bf16, (unsigned short)r);
}

// ---------------------------------------------------------------------------
// Embedding gather: x[t, :] = wte[idx[t], :]
// ---------------------------------------------------------------------------
__global__ __launch_bounds__(256) void embed_kernel(const int* __restrict__ idx,
                                                    const float* __restrict__ wte,
                                                    float* __restrict__ x) {
    int t = blockIdx.x;
    int v = idx[t];
    const float* src = wte + (size_t)v * 1024;
    float* dst = x + (size_t)t * 1024;
    for (int i = threadIdx.x; i < 1024; i += 256) dst[i] = src[i];
}

// ---------------------------------------------------------------------------
// Transpose + f32->bf16 convert: in[K][N] f32 -> out[N][K] bf16
// ---------------------------------------------------------------------------
__global__ __launch_bounds__(256) void convT_kernel(const float* __restrict__ in,
                                                    __bf16* __restrict__ out,
                                                    int K, int N) {
    __shared__ float tile[32][33];
    int tx = threadIdx.x & 31;
    int ty = threadIdx.x >> 5;              // 0..7
    int nb = blockIdx.x * 32;
    int kb = blockIdx.y * 32;
    #pragma unroll
    for (int i = 0; i < 4; i++) {
        int r = ty + i * 8;
        tile[r][tx] = in[(size_t)(kb + r) * N + nb + tx];
    }
    __syncthreads();
    #pragma unroll
    for (int i = 0; i < 4; i++) {
        int r = ty + i * 8;
        out[(size_t)(nb + r) * K + kb + tx] = f2bf(tile[tx][r]);
    }
}

// ---------------------------------------------------------------------------
// LayerNorm over D=1024, one block (256 thr) per row. outBf16 selects dtype.
// ---------------------------------------------------------------------------
__global__ __launch_bounds__(256) void ln_kernel(const float* __restrict__ x,
                                                 const float* __restrict__ w,
                                                 const float* __restrict__ b,
                                                 void* __restrict__ out,
                                                 int outBf16) {
    __shared__ float red[256];
    int row = blockIdx.x;
    const float* xr = x + (size_t)row * 1024;
    float vals[4];
    float s = 0.f;
    #pragma unroll
    for (int i = 0; i < 4; i++) { vals[i] = xr[threadIdx.x + i * 256]; s += vals[i]; }
    red[threadIdx.x] = s; __syncthreads();
    for (int st = 128; st > 0; st >>= 1) {
        if (threadIdx.x < st) red[threadIdx.x] += red[threadIdx.x + st];
        __syncthreads();
    }
    float mean = red[0] * (1.f / 1024.f);
    __syncthreads();
    float s2 = 0.f;
    #pragma unroll
    for (int i = 0; i < 4; i++) { float d = vals[i] - mean; s2 += d * d; }
    red[threadIdx.x] = s2; __syncthreads();
    for (int st = 128; st > 0; st >>= 1) {
        if (threadIdx.x < st) red[threadIdx.x] += red[threadIdx.x + st];
        __syncthreads();
    }
    float rstd = rsqrtf(red[0] * (1.f / 1024.f) + 1e-5f);
    #pragma unroll
    for (int i = 0; i < 4; i++) {
        int d = threadIdx.x + i * 256;
        float o = (vals[i] - mean) * rstd * w[d] + b[d];
        if (outBf16) ((__bf16*)out)[(size_t)row * 1024 + d] = f2bf(o);
        else         ((float*)out)[(size_t)row * 1024 + d] = o;
    }
}

// ---------------------------------------------------------------------------
// bf16 WMMA GEMM: out[M][N] = epilogue( A[M][K] @ WT[N][K]^T + bias )
// Block tile 128x128, 8 waves (4m x 2n), wave tile 32x64 (2x4 WMMA 16x16x32).
// Double-buffered async staging: issue tile k+1, wait asynccnt<=4 (in-order
// retire guarantees tile k landed), compute tile k while k+1 streams in.
// flags: 1 = bf16 output, 2 = exact GELU, 4 = f32 residual add.
// ---------------------------------------------------------------------------
#define GBM 128
#define GBN 128
#define GBK 32
#define AST 40   // padded LDS stride (bf16 elems): 80B rows, 16B aligned

__global__ __launch_bounds__(256) void gemm_bf16_kernel(
    const __bf16* __restrict__ A, const __bf16* __restrict__ WT,
    const float* __restrict__ bias, const float* __restrict__ res,
    void* __restrict__ out, int M, int N, int K, int flags, float scale) {
    __shared__ __bf16 Asl[2][GBM][AST];   // 2 x 10KB
    __shared__ __bf16 Wsl[2][GBN][AST];   // 2 x 10KB

    const int tid  = threadIdx.x;
    const int lane = tid & 31;
    const int wid  = tid >> 5;
    const int wm   = wid >> 1;       // 0..3 -> 32 rows each
    const int wn   = wid & 1;        // 0..1 -> 64 cols each
    const int m0   = blockIdx.y * GBM;
    const int n0   = blockIdx.x * GBN;
    const int lm   = lane & 15;
    const int hiq  = lane >> 4;

    const v8f vzero = {};
    v8f acc[2][4];
    #pragma unroll
    for (int a = 0; a < 2; a++)
        #pragma unroll
        for (int b = 0; b < 4; b++) acc[a][b] = vzero;

    const int sr   = tid >> 1;       // 0..127 tile row (for both A and W tiles)
    const int sseg = tid & 1;        // two 32B halves of the 64B row slice

    const __bf16* gabase = A  + (size_t)(m0 + sr) * K + sseg * 16;
    const __bf16* gwbase = WT + (size_t)(n0 + sr) * K + sseg * 16;

    // stage one K-slice (A + W) of tile `kbase` into buffer `buf`
    auto issue_tile = [&](int buf, int kbase) {
#if USE_ASYNC_LDS
        const __bf16* ga = gabase + kbase;
        const __bf16* gw = gwbase + kbase;
        __bf16* la = &Asl[buf][sr][sseg * 16];
        __bf16* lw = &Wsl[buf][sr][sseg * 16];
        ASYNC_CP16(ga,     la);
        ASYNC_CP16(ga + 8, la + 8);
        ASYNC_CP16(gw,     lw);
        ASYNC_CP16(gw + 8, lw + 8);
#else
        const __bf16* ga = gabase + kbase;
        const __bf16* gw = gwbase + kbase;
        uint4 av0 = ((const uint4*)ga)[0];
        uint4 av1 = ((const uint4*)ga)[1];
        uint4 wv0 = ((const uint4*)gw)[0];
        uint4 wv1 = ((const uint4*)gw)[1];
        __bf16* la = &Asl[buf][sr][sseg * 16];
        __bf16* lw = &Wsl[buf][sr][sseg * 16];
        *(uint4*)la       = av0;
        *(uint4*)(la + 8) = av1;
        *(uint4*)lw       = wv0;
        *(uint4*)(lw + 8) = wv1;
#endif
    };

    auto compute_tile = [&](int buf) {
        FragU af[2];
        #pragma unroll
        for (int mt = 0; mt < 2; mt++) {
            const unsigned* arow = (const unsigned*)&Asl[buf][wm * 32 + mt * 16 + lm][0];
            #pragma unroll
            for (int j = 0; j < 8; j++) {
                int k0 = 2 * j + hiq * 16;   // A-frag k pairing
                af[mt].u[j] = arow[k0 >> 1];
            }
        }
        #pragma unroll
        for (int nt = 0; nt < 4; nt++) {
            FragU bfr;
            const unsigned* wrow = (const unsigned*)&Wsl[buf][wn * 64 + nt * 16 + lm][0];
            #pragma unroll
            for (int j = 0; j < 8; j++)                            // B-frag k = 2j + hi*16
                bfr.u[j] = wrow[j + hiq * 8];
            #pragma unroll
            for (int mt = 0; mt < 2; mt++)
                acc[mt][nt] = __builtin_amdgcn_wmma_f32_16x16x32_bf16(
                    false, af[mt].v, false, bfr.v, (short)0, acc[mt][nt], false, false);
        }
    };

#if USE_ASYNC_LDS
    issue_tile(0, 0);                         // prologue: prime buffer 0
    for (int kb = 0; kb < K; kb += GBK) {
        const int cur = (kb >> 5) & 1;
        if (kb + GBK < K) {
            issue_tile(cur ^ 1, kb + GBK);    // stream next tile into other buffer
            if (kb + 2 * GBK < K) {           // warm L2 two tiles ahead
                __builtin_prefetch(gabase + kb + 2 * GBK, 0, 3);
                __builtin_prefetch(gwbase + kb + 2 * GBK, 0, 3);
            }
            __builtin_amdgcn_s_wait_asynccnt(4);   // oldest 4 (tile kb) retired
        } else {
            __builtin_amdgcn_s_wait_asynccnt(0);   // last tile: drain
        }
        __syncthreads();                      // all waves' tile-kb loads landed
        compute_tile(cur);
        __syncthreads();                      // reads done before buf reuse at kb+2
    }
#else
    for (int kb = 0; kb < K; kb += GBK) {
        issue_tile(0, kb);
        if (kb + GBK < K) {
            __builtin_prefetch(gabase + kb + GBK, 0, 3);
            __builtin_prefetch(gwbase + kb + GBK, 0, 3);
        }
        __syncthreads();
        compute_tile(0);
        __syncthreads();
    }
#endif

    #pragma unroll
    for (int mt = 0; mt < 2; mt++)
        #pragma unroll
        for (int nt = 0; nt < 4; nt++) {
            int col = n0 + wn * 64 + nt * 16 + lm;
            float bv = bias ? bias[col] : 0.f;
            #pragma unroll
            for (int r = 0; r < 8; r++) {
                int row = m0 + wm * 32 + mt * 16 + r + hiq * 8;
                float v = (acc[mt][nt][r] + bv) * scale;
                if (flags & 2) v = 0.5f * v * (1.0f + erff(v * 0.70710678118654752f));
                if (flags & 4) v += res[(size_t)row * N + col];
                if (flags & 1) ((__bf16*)out)[(size_t)row * N + col] = f2bf(v);
                else           ((float*)out)[(size_t)row * N + col] = v;
            }
        }
}

// ---------------------------------------------------------------------------
// Flash attention w/ ALiBi + causal, 1 wave per (head, 16-query tile).
// q already scaled by 1/sqrt(64). kv layout: [T][2048] bf16, K at col h*64,
// V at col 1024 + h*64. Scores via WMMA, online softmax via half-wave
// shuffles, P relayout through LDS to an A-fragment, att@V via WMMA.
// ---------------------------------------------------------------------------
__global__ __launch_bounds__(32) void attn_kernel(const __bf16* __restrict__ qb,
                                                  const __bf16* __restrict__ kvb,
                                                  __bf16* __restrict__ yb) {
    __shared__ __bf16 Pl[16][32];
    const int lane = threadIdx.x & 31;
    const int lm   = lane & 15;
    const int hiq  = lane >> 4;
    const int h    = blockIdx.x;    // 0..15
    const int qt   = blockIdx.y;    // 0..127
    const float slope = exp2f(-0.5f * (float)(h + 1));

    FragU qa[2];
    {
        int tq = qt * 16 + lm;
        const __bf16* qrow = qb + (size_t)tq * 1024 + h * 64;
        #pragma unroll
        for (int dblk = 0; dblk < 2; dblk++)
            #pragma unroll
            for (int j = 0; j < 8; j++) {
                int d = dblk * 32 + 2 * j + hiq * 16;
                qa[dblk].u[j] = *(const unsigned*)(qrow + d);
            }
    }

    const v8f vzero = {};
    float mrun[8], lrun[8];
    v8f acc[4];
    #pragma unroll
    for (int r = 0; r < 8; r++) { mrun[r] = -1e30f; lrun[r] = 0.f; }
    #pragma unroll
    for (int t = 0; t < 4; t++) acc[t] = vzero;

    const int tqmax = qt * 16 + 15;
    for (int kb = 0; kb <= tqmax; kb += 32) {
        v8f S[2];
        #pragma unroll
        for (int nt = 0; nt < 2; nt++) {
            int tk = kb + nt * 16 + lm;
            const __bf16* krow = kvb + (size_t)tk * 2048 + h * 64;
            FragU kf0, kf1;
            #pragma unroll
            for (int j = 0; j < 8; j++) {
                int d = 2 * j + hiq * 16;
                kf0.u[j] = *(const unsigned*)(krow + d);
                kf1.u[j] = *(const unsigned*)(krow + 32 + d);
            }
            v8f s = vzero;
            s = __builtin_amdgcn_wmma_f32_16x16x32_bf16(false, qa[0].v, false, kf0.v,
                                                        (short)0, s, false, false);
            s = __builtin_amdgcn_wmma_f32_16x16x32_bf16(false, qa[1].v, false, kf1.v,
                                                        (short)0, s, false, false);
            S[nt] = s;
        }

        float pnew[2][8];
        #pragma unroll
        for (int r = 0; r < 8; r++) {
            int tqr = qt * 16 + r + hiq * 8;
            float best = -1e30f;
            #pragma unroll
            for (int nt = 0; nt < 2; nt++) {
                int tk = kb + nt * 16 + lm;
                float s = S[nt][r];
                s = (tk <= tqr) ? (s + slope * (float)(tk - tqr)) : -1e30f;
                pnew[nt][r] = s;
                best = fmaxf(best, s);
            }
            for (int msk = 1; msk < 16; msk <<= 1)
                best = fmaxf(best, __shfl_xor(best, msk, 32));   // half-wave row max
            float mnew = fmaxf(mrun[r], best);
            float alpha = expf(mrun[r] - mnew);
            mrun[r] = mnew;
            float rs = 0.f;
            #pragma unroll
            for (int nt = 0; nt < 2; nt++) {
                float p = expf(pnew[nt][r] - mnew);
                pnew[nt][r] = p;
                rs += p;
            }
            for (int msk = 1; msk < 16; msk <<= 1)
                rs += __shfl_xor(rs, msk, 32);                   // half-wave row sum
            lrun[r] = lrun[r] * alpha + rs;
            #pragma unroll
            for (int t = 0; t < 4; t++) acc[t][r] *= alpha;      // rescale O rows
        }

        // relayout P: C-layout -> LDS [m][k] -> A-fragment
        #pragma unroll
        for (int nt = 0; nt < 2; nt++)
            #pragma unroll
            for (int r = 0; r < 8; r++)
                Pl[r + hiq * 8][nt * 16 + lm] = f2bf(pnew[nt][r]);
        __syncthreads();
        FragU pf;
        {
            const unsigned* prow = (const unsigned*)&Pl[lm][0];
            #pragma unroll
            for (int j = 0; j < 8; j++) {
                int k0 = 2 * j + hiq * 16;
                pf.u[j] = prow[k0 >> 1];
            }
        }
        __syncthreads();

        #pragma unroll
        for (int ot = 0; ot < 4; ot++) {
            FragU vf;
            #pragma unroll
            for (int j = 0; j < 8; j++) {
                int k0 = kb + 2 * j + hiq * 16;
                const __bf16* v0 = kvb + (size_t)k0 * 2048 + 1024 + h * 64 + ot * 16 + lm;
                unsigned lo = *(const unsigned short*)v0;
                unsigned hi = *(const unsigned short*)(v0 + 2048);
                vf.u[j] = lo | (hi << 16);
            }
            acc[ot] = __builtin_amdgcn_wmma_f32_16x16x32_bf16(false, pf.v, false, vf.v,
                                                              (short)0, acc[ot], false, false);
        }
    }

    #pragma unroll
    for (int ot = 0; ot < 4; ot++)
        #pragma unroll
        for (int r = 0; r < 8; r++) {
            int t = qt * 16 + r + hiq * 8;
            float o = acc[ot][r] / lrun[r];
            yb[(size_t)t * 1024 + h * 64 + ot * 16 + lm] = f2bf(o);
        }
}

// ---------------------------------------------------------------------------
// Tied head GEMV: logits[v] = dot(xln, wte[v])  (bandwidth-bound, f32)
// ---------------------------------------------------------------------------
__global__ __launch_bounds__(256) void head_kernel(const float* __restrict__ xln,
                                                   const float* __restrict__ wte,
                                                   float* __restrict__ out) {
    __shared__ float xs[1024];
    for (int i = threadIdx.x; i < 1024; i += 256) xs[i] = xln[i];
    __syncthreads();
    int lane = threadIdx.x & 31;
    int wv   = threadIdx.x >> 5;
    int row  = blockIdx.x * 8 + wv;
    const float* wr = wte + (size_t)row * 1024;
    float s = 0.f;
    for (int d = lane; d < 1024; d += 32) s += xs[d] * wr[d];
    for (int m = 1; m < 32; m <<= 1) s += __shfl_xor(s, m, 32);
    if (lane == 0) out[row] = s;
}

// ---------------------------------------------------------------------------
// Host-side orchestration
// ---------------------------------------------------------------------------
extern "C" void kernel_launch(void* const* d_in, const int* in_sizes, int n_in,
                              void* d_out, int out_size, void* d_ws, size_t ws_size,
                              hipStream_t stream) {
    (void)in_sizes; (void)n_in; (void)out_size; (void)ws_size;
    const int*   idx    = (const int*)  d_in[0];
    const float* wte    = (const float*)d_in[1];
    const float* ln1_w  = (const float*)d_in[2];
    const float* ln1_b  = (const float*)d_in[3];
    const float* wq     = (const float*)d_in[4];
    const float* bq     = (const float*)d_in[5];
    const float* wkv    = (const float*)d_in[6];
    const float* bkv    = (const float*)d_in[7];
    const float* wo     = (const float*)d_in[8];
    const float* bo     = (const float*)d_in[9];
    const float* ln2_w  = (const float*)d_in[10];
    const float* ln2_b  = (const float*)d_in[11];
    const float* wfc    = (const float*)d_in[12];
    const float* bfc    = (const float*)d_in[13];
    const float* wproj  = (const float*)d_in[14];
    const float* bproj  = (const float*)d_in[15];
    const float* lnf_w  = (const float*)d_in[16];
    const float* lnf_b  = (const float*)d_in[17];

    const int T = 2048, D = 1024;

    char* p = (char*)d_ws;
    float*  x    = (float*)p;   p += (size_t)T * D * 4;         // residual stream, f32
    __bf16* hbuf = (__bf16*)p;  p += (size_t)T * D * 2;         // LN output
    __bf16* qb   = (__bf16*)p;  p += (size_t)T * D * 2;         // Q (prescaled)
    __bf16* kvb  = (__bf16*)p;  p += (size_t)T * 2 * D * 2;     // K|V
    __bf16* yb   = (__bf16*)p;  p += (size_t)T * D * 2;         // attn out
    __bf16* mfc  = (__bf16*)p;  p += (size_t)T * 4 * D * 2;     // MLP hidden
    __bf16* wT   = (__bf16*)p;  p += (size_t)4 * D * D * 2;     // 8MB transposed-weight scratch
    float*  xln  = (float*)p;   p += 4096;                      // final LN of last token

    embed_kernel<<<T, 256, 0, stream>>>(idx, wte, x);

    for (int l = 0; l < 4; l++) {
        const float* Wq  = wq    + (size_t)l * D * D;
        const float* Wkv = wkv   + (size_t)l * D * 2 * D;
        const float* Wo  = wo    + (size_t)l * D * D;
        const float* Wfc = wfc   + (size_t)l * D * 4 * D;
        const float* Wpr = wproj + (size_t)l * 4 * D * D;

        // ---- attention ----
        ln_kernel<<<T, 256, 0, stream>>>(x, ln1_w + l * D, ln1_b + l * D, hbuf, 1);

        convT_kernel<<<dim3(D / 32, D / 32), 256, 0, stream>>>(Wq, wT, D, D);
        gemm_bf16_kernel<<<dim3(D / GBN, T / GBM), 256, 0, stream>>>(
            hbuf, wT, bq + l * D, nullptr, qb, T, D, D, /*bf16*/1, 0.125f);

        convT_kernel<<<dim3(2 * D / 32, D / 32), 256, 0, stream>>>(Wkv, wT, D, 2 * D);
        gemm_bf16_kernel<<<dim3(2 * D / GBN, T / GBM), 256, 0, stream>>>(
            hbuf, wT, bkv + (size_t)l * 2 * D, nullptr, kvb, T, 2 * D, D, 1, 1.0f);

        attn_kernel<<<dim3(16, T / 16), 32, 0, stream>>>(qb, kvb, yb);

        convT_kernel<<<dim3(D / 32, D / 32), 256, 0, stream>>>(Wo, wT, D, D);
        gemm_bf16_kernel<<<dim3(D / GBN, T / GBM), 256, 0, stream>>>(
            yb, wT, bo + l * D, x, x, T, D, D, /*residual,f32*/4, 1.0f);

        // ---- MLP ----
        ln_kernel<<<T, 256, 0, stream>>>(x, ln2_w + l * D, ln2_b + l * D, hbuf, 1);

        convT_kernel<<<dim3(4 * D / 32, D / 32), 256, 0, stream>>>(Wfc, wT, D, 4 * D);
        gemm_bf16_kernel<<<dim3(4 * D / GBN, T / GBM), 256, 0, stream>>>(
            hbuf, wT, bfc + (size_t)l * 4 * D, nullptr, mfc, T, 4 * D, D, /*bf16+gelu*/3, 1.0f);

        convT_kernel<<<dim3(D / 32, 4 * D / 32), 256, 0, stream>>>(Wpr, wT, 4 * D, D);
        gemm_bf16_kernel<<<dim3(D / GBN, T / GBM), 256, 0, stream>>>(
            mfc, wT, bproj + l * D, x, x, T, D, 4 * D, 4, 1.0f);
    }

    ln_kernel<<<1, 256, 0, stream>>>(x + (size_t)(T - 1) * D, lnf_w, lnf_b, xln, 0);
    head_kernel<<<32000 / 8, 256, 0, stream>>>(xln, wte, (float*)d_out);
}
